// GCN2_21827023798529
// MI455X (gfx1250) — matrix operand order repaired and checked by reading, not compiled, not run on hardware
//
#include <hip/hip_runtime.h>
#include <cmath>

typedef __attribute__((ext_vector_type(16))) _Float16 v16h;
typedef __attribute__((ext_vector_type(8)))  float    v8f;

#define N_NODES   50000
#define N_EDGES   800000
#define NFEAT     512
#define HID       96
#define NCLASS    64
#define NUM_LAYERS 8
#define ALPHA_C   0.1f

// ---------------------------------------------------------------------------
// A-fragment swizzle: (m, k) -> f16 element offset in a pre-swizzled tile.
// Layout per 32-wide k-tile: [lane(32)][elem(16)], lane = m + 16*hi,
// where for k' = k%32:  hi = (k'>>3)&1,  elem = (k'&7) + 8*(k'>=16).
// A WMMA lane then reads its 16 elements as one contiguous 32-byte chunk.
// ---------------------------------------------------------------------------
__device__ __forceinline__ int swzA(int m, int k) {
  int kt = k >> 5, kp = k & 31;
  int hi = (kp >> 3) & 1;
  int e  = (kp & 7) + ((kp >= 16) ? 8 : 0);
  return ((kt << 5) + (m & 15) + (hi << 4)) * 16 + e;
}

// ---------------------------------------------------------------------------
// Weight pre-pack into B-fragment order (f32 -> f16, one-time per launch).
// Fragment index = ((kt*nct + ct)*32 + lane)*16 + e, source k = kt*32 +
// (lane<16 ? 0 : 8) + (e&7) + (e>=8 ? 16 : 0), n = ct*16 + (lane&15).
// ---------------------------------------------------------------------------
__device__ __forceinline__ void prepackB(const float* __restrict__ W,
                                         _Float16* __restrict__ F,
                                         int idx, int C, int nct) {
  int e    = idx & 15;
  int lane = (idx >> 4) & 31;
  int rest = idx >> 9;
  int ct   = rest % nct;
  int kt   = rest / nct;
  int k = kt * 32 + ((lane < 16) ? 0 : 8) + (e & 7) + ((e >= 8) ? 16 : 0);
  int n = ct * 16 + (lane & 15);
  F[idx] = (_Float16)W[k * C + n];
}

__global__ void cvt_weights_kernel(const float* __restrict__ Win,
                                   const float* __restrict__ Wconv,
                                   const float* __restrict__ Wout,
                                   _Float16* __restrict__ fWin,
                                   _Float16* __restrict__ fWconv,
                                   _Float16* __restrict__ fWout) {
  int i = blockIdx.x * blockDim.x + threadIdx.x;
  if (i < NFEAT * HID) prepackB(Win, fWin, i, HID, HID / 16);
  if (i < NUM_LAYERS * HID * HID) {
    int l = i / (HID * HID);
    int j = i - l * (HID * HID);
    prepackB(Wconv + (size_t)l * HID * HID, fWconv + (size_t)l * HID * HID,
             j, HID, HID / 16);
  }
  if (i < HID * NCLASS) prepackB(Wout, fWout, i, NCLASS, NCLASS / 16);
}

// ---------------------------------------------------------------------------
// CSR-by-destination build: counts -> exclusive scan -> fill
// ---------------------------------------------------------------------------
__global__ void zero_counts_kernel(int* __restrict__ counts) {
  int i = blockIdx.x * blockDim.x + threadIdx.x;
  if (i < N_NODES) counts[i] = 0;
}

__global__ void count_kernel(const int* __restrict__ dst, int* __restrict__ counts) {
  int e = blockIdx.x * blockDim.x + threadIdx.x;
  if (e < N_EDGES) atomicAdd(&counts[dst[e]], 1);
}

__global__ void scan_blocks_kernel(const int* __restrict__ counts,
                                   int* __restrict__ excl, int* __restrict__ bsums) {
  __shared__ int sh[256];
  int tid = threadIdx.x;
  int i = blockIdx.x * 256 + tid;
  int v = (i < N_NODES) ? counts[i] : 0;
  sh[tid] = v;
  __syncthreads();
  for (int off = 1; off < 256; off <<= 1) {
    int t = (tid >= off) ? sh[tid - off] : 0;
    __syncthreads();
    sh[tid] += t;
    __syncthreads();
  }
  if (i < N_NODES) excl[i] = sh[tid] - v;   // exclusive within block
  if (tid == 255) bsums[blockIdx.x] = sh[255];
}

__global__ void scan_sums_kernel(const int* __restrict__ bsums,
                                 int* __restrict__ boffs, int nblocks) {
  __shared__ int sh[256];
  int tid = threadIdx.x;
  int v = (tid < nblocks) ? bsums[tid] : 0;
  sh[tid] = v;
  __syncthreads();
  for (int off = 1; off < 256; off <<= 1) {
    int t = (tid >= off) ? sh[tid - off] : 0;
    __syncthreads();
    sh[tid] += t;
    __syncthreads();
  }
  boffs[tid] = sh[tid] - v;                 // exclusive block offsets
}

__global__ void add_offsets_kernel(int* __restrict__ row_ptr,
                                   const int* __restrict__ boffs,
                                   int* __restrict__ cursor) {
  int i = blockIdx.x * blockDim.x + threadIdx.x;
  if (i < N_NODES) {
    int v = row_ptr[i] + boffs[i >> 8];
    row_ptr[i] = v;
    cursor[i] = v;
  }
  if (i == 0) row_ptr[N_NODES] = N_EDGES;
}

__global__ void fill_csr_kernel(const int* __restrict__ src, const int* __restrict__ dst,
                                int* __restrict__ cursor, int* __restrict__ csr_src) {
  int e = blockIdx.x * blockDim.x + threadIdx.x;
  if (e < N_EDGES) {
    int p = atomicAdd(&cursor[dst[e]], 1);
    csr_src[p] = src[e];
  }
}

// ---------------------------------------------------------------------------
// Input GEMM: h = relu(x @ W_in + b_in), also x0 = h
// Block: 192 threads = 6 waves; 16 rows x 96 cols per block, K = 512.
// A tile staged pre-swizzled in LDS; fragments read as contiguous 32B.
// ---------------------------------------------------------------------------
__global__ __launch_bounds__(192)
void gemm_in_kernel(const float* __restrict__ x, const _Float16* __restrict__ Wf,
                    const float* __restrict__ bias, float* __restrict__ h,
                    float* __restrict__ x0) {
  __shared__ __align__(32) _Float16 xt[16 * NFEAT];
  const int tid = threadIdx.x;
  const int lane = tid & 31, wave = tid >> 5;
  const int row0 = blockIdx.x * 16;

  for (int i = tid; i < 16 * NFEAT; i += 192) {
    int m = i >> 9, k = i & (NFEAT - 1);
    xt[swzA(m, k)] = (_Float16)x[(size_t)(row0 + m) * NFEAT + k];
  }
  __syncthreads();

  const int n = lane & 15;
  v8f acc = {};
#pragma unroll 4
  for (int kt = 0; kt < NFEAT / 32; ++kt) {
    v16h a = *(const v16h*)&xt[((kt << 5) + lane) << 4];
    v16h b = *(const v16h*)(Wf + (size_t)(((kt * (HID / 16) + wave) << 5) + lane) * 16);
    acc = __builtin_amdgcn_wmma_f32_16x16x32_f16(false, a, false, b,
                                                 (short)0, acc, false, false);
  }

  const int rb  = (lane < 16) ? 0 : 8;
  const int col = wave * 16 + n;
#pragma unroll
  for (int r = 0; r < 8; ++r) {
    int row = row0 + rb + r;
    float v = acc[r] + bias[col];
    v = fmaxf(v, 0.0f);
    h[(size_t)row * HID + col]  = v;
    x0[(size_t)row * HID + col] = v;
  }
}

// ---------------------------------------------------------------------------
// Fused layer: CSR gather-aggregate (float3 per lane) into pre-swizzled LDS
// (hc = 0.9*agg + 0.1*x0 as f16), then per-wave 16x16 WMMA over K=96,
// h_out = relu((1-b)*hc + b*(hc@W)).  Block: 192 threads = 6 waves, 16 nodes.
// ---------------------------------------------------------------------------
__global__ __launch_bounds__(192)
void layer_kernel(const float* __restrict__ h_in, float* __restrict__ h_out,
                  const float* __restrict__ x0, const int* __restrict__ row_ptr,
                  const int* __restrict__ csr_src, const _Float16* __restrict__ Wf,
                  float beta) {
  __shared__ __align__(32) _Float16 hc[16 * HID];
  const int tid = threadIdx.x, lane = tid & 31, wave = tid >> 5;
  const int node0 = blockIdx.x * 16;

  // gather aggregation: one wave per node, lane owns features {3l, 3l+1, 3l+2}
  for (int r = wave; r < 16; r += 6) {
    const int node = node0 + r;
    const int beg = row_ptr[node], end = row_ptr[node + 1];
    float s0 = 0.f, s1 = 0.f, s2 = 0.f;
    for (int idx = beg; idx < end; ++idx) {
      const float3 v = ((const float3*)(h_in + (size_t)csr_src[idx] * HID))[lane];
      s0 += v.x; s1 += v.y; s2 += v.z;
    }
    const float3 xv = ((const float3*)(x0 + (size_t)node * HID))[lane];
    const int f = 3 * lane;
    hc[swzA(r, f)]     = (_Float16)((1.f - ALPHA_C) * s0 + ALPHA_C * xv.x);
    hc[swzA(r, f + 1)] = (_Float16)((1.f - ALPHA_C) * s1 + ALPHA_C * xv.y);
    hc[swzA(r, f + 2)] = (_Float16)((1.f - ALPHA_C) * s2 + ALPHA_C * xv.z);
  }
  __syncthreads();

  const int n = lane & 15;
  v8f acc = {};
#pragma unroll
  for (int kt = 0; kt < HID / 32; ++kt) {
    v16h a = *(const v16h*)&hc[((kt << 5) + lane) << 4];
    v16h b = *(const v16h*)(Wf + (size_t)(((kt * (HID / 16) + wave) << 5) + lane) * 16);
    acc = __builtin_amdgcn_wmma_f32_16x16x32_f16(false, a, false, b,
                                                 (short)0, acc, false, false);
  }

  const int rb  = (lane < 16) ? 0 : 8;
  const int col = wave * 16 + n;
#pragma unroll
  for (int r = 0; r < 8; ++r) {
    int lr = rb + r;
    float hcv = (float)hc[swzA(lr, col)];
    float v = (1.f - beta) * hcv + beta * acc[r];
    h_out[(size_t)(node0 + lr) * HID + col] = fmaxf(v, 0.f);
  }
}

// ---------------------------------------------------------------------------
// Output GEMM: logits = h @ W_out + b_out  (50000 x 96 x 64)
// Block: 128 threads = 4 waves; 16 rows x 64 cols, K = 96.
// ---------------------------------------------------------------------------
__global__ __launch_bounds__(128)
void gemm_out_kernel(const float* __restrict__ h, const _Float16* __restrict__ Wf,
                     const float* __restrict__ bias, float* __restrict__ out) {
  __shared__ __align__(32) _Float16 ht[16 * HID];
  const int tid = threadIdx.x, lane = tid & 31, wave = tid >> 5;
  const int row0 = blockIdx.x * 16;

  for (int i = tid; i < 16 * HID; i += 128) {
    int m = i / HID, k = i - m * HID;
    ht[swzA(m, k)] = (_Float16)h[(size_t)row0 * HID + i];
  }
  __syncthreads();

  const int n = lane & 15;
  v8f acc = {};
#pragma unroll
  for (int kt = 0; kt < HID / 32; ++kt) {
    v16h a = *(const v16h*)&ht[((kt << 5) + lane) << 4];
    v16h b = *(const v16h*)(Wf + (size_t)(((kt * (NCLASS / 16) + wave) << 5) + lane) * 16);
    acc = __builtin_amdgcn_wmma_f32_16x16x32_f16(false, a, false, b,
                                                 (short)0, acc, false, false);
  }

  const int rb  = (lane < 16) ? 0 : 8;
  const int col = wave * 16 + n;
#pragma unroll
  for (int r = 0; r < 8; ++r)
    out[(size_t)(row0 + rb + r) * NCLASS + col] = acc[r] + bias[col];
}

// ---------------------------------------------------------------------------
// Row-wise log-softmax over 64 classes, in place. One wave per row (wave32):
// each lane owns columns {lane, lane+32}; reductions via 32-lane shuffles.
// ---------------------------------------------------------------------------
__global__ __launch_bounds__(256)
void log_softmax_kernel(float* __restrict__ out) {
  int gw = (blockIdx.x * blockDim.x + threadIdx.x) >> 5;
  int lane = threadIdx.x & 31;
  if (gw >= N_NODES) return;
  float* p = out + (size_t)gw * NCLASS;
  float v0 = p[lane], v1 = p[lane + 32];
  float m = fmaxf(v0, v1);
  for (int off = 16; off > 0; off >>= 1) m = fmaxf(m, __shfl_xor(m, off));
  float s = expf(v0 - m) + expf(v1 - m);
  for (int off = 16; off > 0; off >>= 1) s += __shfl_xor(s, off);
  float lse = m + logf(s);
  p[lane]      = v0 - lse;
  p[lane + 32] = v1 - lse;
}

// ---------------------------------------------------------------------------
// Host launch
// ---------------------------------------------------------------------------
extern "C" void kernel_launch(void* const* d_in, const int* in_sizes, int n_in,
                              void* d_out, int out_size, void* d_ws, size_t ws_size,
                              hipStream_t stream) {
  (void)in_sizes; (void)n_in; (void)out_size; (void)ws_size;
  const float* x      = (const float*)d_in[0];
  const int*   ei     = (const int*)d_in[1];
  const float* W_in   = (const float*)d_in[2];
  const float* b_in   = (const float*)d_in[3];
  const float* conv_W = (const float*)d_in[4];
  const float* W_out  = (const float*)d_in[5];
  const float* b_out  = (const float*)d_in[6];
  float* out = (float*)d_out;
  const int* src = ei;
  const int* dst = ei + N_EDGES;

  char* ws = (char*)d_ws;
  size_t off = 0;
  auto carve = [&](size_t bytes) -> void* {
    void* p = ws + off;
    off = (off + bytes + 255) & ~(size_t)255;
    return p;
  };
  _Float16* fWin   = (_Float16*)carve((size_t)NFEAT * HID * 2);
  _Float16* fWconv = (_Float16*)carve((size_t)NUM_LAYERS * HID * HID * 2);
  _Float16* fWout  = (_Float16*)carve((size_t)HID * NCLASS * 2);
  float* h_a    = (float*)carve((size_t)N_NODES * HID * 4);
  float* h_b    = (float*)carve((size_t)N_NODES * HID * 4);
  float* x0     = (float*)carve((size_t)N_NODES * HID * 4);
  int* counts   = (int*)carve((size_t)N_NODES * 4);
  int* row_ptr  = (int*)carve((size_t)(N_NODES + 1) * 4);
  int* cursor   = (int*)carve((size_t)N_NODES * 4);
  int* bsums    = (int*)carve(256 * 4);
  int* boffs    = (int*)carve(256 * 4);
  int* csr_src  = (int*)carve((size_t)N_EDGES * 4);

  const int SCAN_BLOCKS = (N_NODES + 255) / 256;   // 196
  const int EDGE_BLOCKS = (N_EDGES + 255) / 256;   // 3125
  const int TILE_BLOCKS = N_NODES / 16;            // 3125 (50000 % 16 == 0)

  cvt_weights_kernel<<<(NUM_LAYERS * HID * HID + 255) / 256, 256, 0, stream>>>(
      W_in, conv_W, W_out, fWin, fWconv, fWout);

  zero_counts_kernel<<<SCAN_BLOCKS, 256, 0, stream>>>(counts);
  count_kernel<<<EDGE_BLOCKS, 256, 0, stream>>>(dst, counts);
  scan_blocks_kernel<<<SCAN_BLOCKS, 256, 0, stream>>>(counts, row_ptr, bsums);
  scan_sums_kernel<<<1, 256, 0, stream>>>(bsums, boffs, SCAN_BLOCKS);
  add_offsets_kernel<<<SCAN_BLOCKS, 256, 0, stream>>>(row_ptr, boffs, cursor);
  fill_csr_kernel<<<EDGE_BLOCKS, 256, 0, stream>>>(src, dst, cursor, csr_src);

  gemm_in_kernel<<<TILE_BLOCKS, 192, 0, stream>>>(x, fWin, b_in, h_a, x0);

  float* hin = h_a;
  float* hout = h_b;
  for (int l = 0; l < NUM_LAYERS; ++l) {
    float beta = logf(0.5f / (float)(l + 1) + 1.0f);
    layer_kernel<<<TILE_BLOCKS, 192, 0, stream>>>(
        hin, hout, x0, row_ptr, csr_src, fWconv + (size_t)l * HID * HID, beta);
    float* t = hin; hin = hout; hout = t;
  }

  gemm_out_kernel<<<TILE_BLOCKS, 128, 0, stream>>>(hin, fWout, b_out, out);
  log_softmax_kernel<<<(N_NODES * 32 + 255) / 256, 256, 0, stream>>>(out);
}
